// HRM_ACTV1_Inner_36799279792523
// MI455X (gfx1250) — matrix operand order repaired and checked by reading, not compiled, not run on hardware
//
#include <hip/hip_runtime.h>
#include <math.h>

// ---------------------------------------------------------------------------
// Transformer block for MI455X (gfx1250, wave32).
//   - all GEMM-shaped math on v_wmma_f32_16x16x32_f16
//   - f16 k-major weights; GEMM B-tiles stage via global_load_async_to_lds_b128
//     (ASYNCcnt), double-buffered LDS ping-pong to overlap staging with WMMA
//   - barrier-free flash attention with pre-transposed f16 K/V + prefetch
// ---------------------------------------------------------------------------

typedef __attribute__((ext_vector_type(16))) _Float16 v16h;
typedef __attribute__((ext_vector_type(8)))  _Float16 v8h;
typedef __attribute__((ext_vector_type(8)))  float    v8f;

#define WMMA_F16(a, b, c) \
  __builtin_amdgcn_wmma_f32_16x16x32_f16(false, (a), false, (b), (short)0, (c), false, false)

// ---- CDNA5 async global->LDS copy (ISA 10.x / 15.18, ASYNCcnt) -------------
__device__ __forceinline__ void async_copy_b128(unsigned lds_off, const void* gaddr) {
  asm volatile("global_load_async_to_lds_b128 %0, %1, off"
               :: "v"(lds_off), "v"(gaddr) : "memory");
}
__device__ __forceinline__ void wait_async0() {
  asm volatile("s_wait_asynccnt 0x0" ::: "memory");
}
__device__ __forceinline__ unsigned lds_off(const void* p) {
  return (unsigned)(unsigned long long)p;   // low 32 bits of flat addr = LDS offset
}

// ---- fragment loaders (wave32 layouts per CDNA5 ISA 7.12.2) ----------------
// A (16x32 f16): lane l holds row m = l%16.
//   lanes 0-15 : K = {0..7, 16..23};  lanes 16-31 : K = {8..15, 24..31}
__device__ __forceinline__ v16h load_a_f32(const float* base, int stride, int lane) {
  const int m = lane & 15, hi = lane >> 4;
  const float* r = base + (size_t)m * stride;
  v16h a;
#pragma unroll
  for (int i = 0; i < 8; ++i) {
    a[i]     = (_Float16)r[hi * 8 + i];
    a[i + 8] = (_Float16)r[16 + hi * 8 + i];
  }
  return a;
}

__device__ __forceinline__ v16h load_a_f16(const _Float16* base, int stride, int lane) {
  const int m = lane & 15, hi = lane >> 4;
  const _Float16* r = base + m * stride + hi * 8;
  v16h a;
#pragma unroll
  for (int i = 0; i < 8; ++i) {
    a[i]     = r[i];
    a[i + 8] = r[16 + i];
  }
  return a;
}

// B (32x16 f16): lane l holds column n = l%16, K order same as A.
__device__ __forceinline__ v16h load_b_f16(const _Float16* base, int stride, int lane) {
  const int n = lane & 15, hi = lane >> 4;
  const _Float16* c = base + (hi * 8) * stride + n;
  v16h b;
#pragma unroll
  for (int i = 0; i < 8; ++i) {
    b[i]     = c[i * stride];
    b[i + 8] = c[(16 + i) * stride];
  }
  return b;
}

// ---- tile staging helpers ---------------------------------------------------
__device__ __forceinline__ void stage_b_async(const _Float16* Wt, int k0, int n0, int N,
                                              _Float16* dst, int tid) {
  const int bk = tid >> 3, bn = (tid & 7) * 8;   // 32x64 tile, one b128/thread
  async_copy_b128(lds_off(dst + bk * 64 + bn),
                  Wt + (size_t)(k0 + bk) * N + n0 + bn);
}

__device__ __forceinline__ void stage_a(const float* A, int m0, int k0, int K,
                                        _Float16* dst, int tid) {
  const int row = tid >> 1, col = (tid & 1) * 16;
  const float* src = A + (size_t)(m0 + row) * K + k0 + col;
  v8h h0, h1;
#pragma unroll
  for (int i = 0; i < 8; ++i) { h0[i] = (_Float16)src[i]; h1[i] = (_Float16)src[8 + i]; }
  *(v8h*)(dst + row * 32 + col)     = h0;
  *(v8h*)(dst + row * 32 + col + 8) = h1;
}

// ---------------------------------------------------------------------------
// One-time weight transform: W[N][K] f32 -> Wt[K][N] f16 (k-major).
// ---------------------------------------------------------------------------
__global__ void __launch_bounds__(256)
transpose_w_f16(const float* __restrict__ W, _Float16* __restrict__ Wt, int N, int K) {
  __shared__ _Float16 tile[32 * 33];
  const int tid = threadIdx.x;
  const int k0 = blockIdx.x * 32, n0 = blockIdx.y * 32;
  {
    const int r = tid >> 3, c = (tid & 7) * 4;   // r: n-row, c: k-col
    const float* src = W + (size_t)(n0 + r) * K + k0 + c;
#pragma unroll
    for (int i = 0; i < 4; ++i) tile[r * 33 + c + i] = (_Float16)src[i];
  }
  __syncthreads();
  {
    const int r = tid >> 3, c = (tid & 7) * 4;   // r: k-row, c: n-col
    _Float16* dst = Wt + (size_t)(k0 + r) * N + n0 + c;
#pragma unroll
    for (int i = 0; i < 4; ++i) dst[i] = tile[(c + i) * 33 + r];
  }
}

// ---------------------------------------------------------------------------
// GEMM: C[M,N] = A[M,K](f32) @ Wt[K,N](f16,k-major) + bias.  BM=128 BN=64 BK=32.
// 256 thr = 8 waves (4x2), 32x32 out per wave.  Double-buffered LDS: next B
// tile via async copy + next A loads issued before this step's WMMAs.
// ---------------------------------------------------------------------------
__global__ void __launch_bounds__(256)
gemm_bias_wmma(const float* __restrict__ A, const _Float16* __restrict__ Wt,
               const float* __restrict__ bias, float* __restrict__ Cout,
               int M, int N, int K) {
  __shared__ __align__(32) _Float16 As[2][128 * 32];   // [m][k]
  __shared__ __align__(32) _Float16 Bs[2][32 * 64];    // [k][n]

  const int tid = threadIdx.x, lane = tid & 31, wv = tid >> 5;
  const int wm = wv >> 1, wn = wv & 1;
  const int m0 = blockIdx.y * 128, n0 = blockIdx.x * 64;
  const int arow = tid >> 1, acol = (tid & 1) * 16;

  v8f acc[2][2] = {};

  stage_b_async(Wt, 0, n0, N, Bs[0], tid);
  stage_a(A, m0, 0, K, As[0], tid);
  wait_async0();
  __syncthreads();

  int cur = 0;
  for (int k0 = 0; k0 < K; k0 += 32) {
    const bool hn = (k0 + 32) < K;
    float fa[16];
    if (hn) {  // issue next-tile traffic before this step's WMMAs
      stage_b_async(Wt, k0 + 32, n0, N, Bs[cur ^ 1], tid);
      const float* src = A + (size_t)(m0 + arow) * K + (k0 + 32) + acol;
#pragma unroll
      for (int i = 0; i < 16; ++i) fa[i] = src[i];
    }

    v16h a0 = load_a_f16(As[cur] + (wm * 32 + 0) * 32, 32, lane);
    v16h a1 = load_a_f16(As[cur] + (wm * 32 + 16) * 32, 32, lane);
    v16h b0 = load_b_f16(Bs[cur] + wn * 32 + 0, 64, lane);
    v16h b1 = load_b_f16(Bs[cur] + wn * 32 + 16, 64, lane);

    acc[0][0] = WMMA_F16(a0, b0, acc[0][0]);
    acc[0][1] = WMMA_F16(a0, b1, acc[0][1]);
    acc[1][0] = WMMA_F16(a1, b0, acc[1][0]);
    acc[1][1] = WMMA_F16(a1, b1, acc[1][1]);

    if (hn) {
      v8h h0, h1;
#pragma unroll
      for (int i = 0; i < 8; ++i) { h0[i] = (_Float16)fa[i]; h1[i] = (_Float16)fa[8 + i]; }
      *(v8h*)(As[cur ^ 1] + arow * 32 + acol)     = h0;
      *(v8h*)(As[cur ^ 1] + arow * 32 + acol + 8) = h1;
    }
    wait_async0();
    __syncthreads();
    cur ^= 1;
  }

  const int hi = lane >> 4, nl = lane & 15;
#pragma unroll
  for (int mi = 0; mi < 2; ++mi)
#pragma unroll
    for (int ni = 0; ni < 2; ++ni) {
      const unsigned col = (unsigned)(n0 + wn * 32 + ni * 16 + nl);
      const float bv = bias[col];
      const unsigned base = (unsigned)(m0 + wm * 32 + mi * 16 + 8 * hi) * (unsigned)N + col;
#pragma unroll
      for (int r = 0; r < 8; ++r)
        Cout[base + (unsigned)(r * N)] = acc[mi][ni][r] + bv;
    }
}

// ---------------------------------------------------------------------------
// Fused dual GEMM + SwiGLU: H = silu(A@W1t + b1) * (A@W2t + b2)
// ---------------------------------------------------------------------------
__global__ void __launch_bounds__(256)
swiglu_wmma(const float* __restrict__ A,
            const _Float16* __restrict__ W1t, const float* __restrict__ B1,
            const _Float16* __restrict__ W2t, const float* __restrict__ B2,
            float* __restrict__ Hout, int M, int N, int K) {
  __shared__ __align__(32) _Float16 As[2][128 * 32];
  __shared__ __align__(32) _Float16 B1s[2][32 * 64];
  __shared__ __align__(32) _Float16 B2s[2][32 * 64];

  const int tid = threadIdx.x, lane = tid & 31, wv = tid >> 5;
  const int wm = wv >> 1, wn = wv & 1;
  const int m0 = blockIdx.y * 128, n0 = blockIdx.x * 64;
  const int arow = tid >> 1, acol = (tid & 1) * 16;

  v8f acc1[2][2] = {}, acc2[2][2] = {};

  stage_b_async(W1t, 0, n0, N, B1s[0], tid);
  stage_b_async(W2t, 0, n0, N, B2s[0], tid);
  stage_a(A, m0, 0, K, As[0], tid);
  wait_async0();
  __syncthreads();

  int cur = 0;
  for (int k0 = 0; k0 < K; k0 += 32) {
    const bool hn = (k0 + 32) < K;
    float fa[16];
    if (hn) {
      stage_b_async(W1t, k0 + 32, n0, N, B1s[cur ^ 1], tid);
      stage_b_async(W2t, k0 + 32, n0, N, B2s[cur ^ 1], tid);
      const float* src = A + (size_t)(m0 + arow) * K + (k0 + 32) + acol;
#pragma unroll
      for (int i = 0; i < 16; ++i) fa[i] = src[i];
    }

    v16h a0 = load_a_f16(As[cur] + (wm * 32 + 0) * 32, 32, lane);
    v16h a1 = load_a_f16(As[cur] + (wm * 32 + 16) * 32, 32, lane);
    v16h c0 = load_b_f16(B1s[cur] + wn * 32 + 0, 64, lane);
    v16h c1 = load_b_f16(B1s[cur] + wn * 32 + 16, 64, lane);
    v16h d0 = load_b_f16(B2s[cur] + wn * 32 + 0, 64, lane);
    v16h d1 = load_b_f16(B2s[cur] + wn * 32 + 16, 64, lane);

    acc1[0][0] = WMMA_F16(a0, c0, acc1[0][0]);
    acc1[0][1] = WMMA_F16(a0, c1, acc1[0][1]);
    acc1[1][0] = WMMA_F16(a1, c0, acc1[1][0]);
    acc1[1][1] = WMMA_F16(a1, c1, acc1[1][1]);
    acc2[0][0] = WMMA_F16(a0, d0, acc2[0][0]);
    acc2[0][1] = WMMA_F16(a0, d1, acc2[0][1]);
    acc2[1][0] = WMMA_F16(a1, d0, acc2[1][0]);
    acc2[1][1] = WMMA_F16(a1, d1, acc2[1][1]);

    if (hn) {
      v8h h0, h1;
#pragma unroll
      for (int i = 0; i < 8; ++i) { h0[i] = (_Float16)fa[i]; h1[i] = (_Float16)fa[8 + i]; }
      *(v8h*)(As[cur ^ 1] + arow * 32 + acol)     = h0;
      *(v8h*)(As[cur ^ 1] + arow * 32 + acol + 8) = h1;
    }
    wait_async0();
    __syncthreads();
    cur ^= 1;
  }

  const int hi = lane >> 4, nl = lane & 15;
#pragma unroll
  for (int mi = 0; mi < 2; ++mi)
#pragma unroll
    for (int ni = 0; ni < 2; ++ni) {
      const unsigned col = (unsigned)(n0 + wn * 32 + ni * 16 + nl);
      const float bv1 = B1[col], bv2 = B2[col];
      const unsigned base = (unsigned)(m0 + wm * 32 + mi * 16 + 8 * hi) * (unsigned)N + col;
#pragma unroll
      for (int r = 0; r < 8; ++r) {
        const float g1 = acc1[mi][ni][r] + bv1;
        const float g2 = acc2[mi][ni][r] + bv2;
        const float sig = 1.0f / (1.0f + __expf(-g1));
        Hout[base + (unsigned)(r * N)] = g1 * sig * g2;
      }
    }
}

// ---------------------------------------------------------------------------
// RoPE in-place on q,k in qkv [B,T,3*H*D]; one thread owns one head vector.
// ---------------------------------------------------------------------------
__global__ void __launch_bounds__(256)
rope_kernel(float* __restrict__ qkv, const float* __restrict__ cs,
            const float* __restrict__ sn) {
  const int idx = blockIdx.x * blockDim.x + threadIdx.x;  // B*T*H = 65536
  if (idx >= 2 * 2048 * 16) return;
  const int h = idx & 15;
  const int t = (idx >> 4) & 2047;
  const int b = idx >> 15;
  const float* cr = cs + t * 32;
  const float* sr = sn + t * 32;
  float* qp = qkv + ((size_t)(b * 2048 + t)) * 3072 + h * 64;
#pragma unroll
  for (int s = 0; s < 2; ++s) {   // q then k
    float* p = qp + s * 1024;
    float u[64];
#pragma unroll
    for (int i = 0; i < 64; ++i) u[i] = p[i];
#pragma unroll
    for (int j = 0; j < 32; ++j) {
      p[j]      = u[2 * j] * cr[j] - u[2 * j + 1] * sr[j];
      p[32 + j] = u[2 * j] * sr[j] + u[2 * j + 1] * cr[j];
    }
  }
}

// ---------------------------------------------------------------------------
// K/V prep: Kt[b,h,d,t] f16 (transposed, B-frag ready for QK^T) and
// Vg[b,h,t,d] f16 (B-frag ready for PV).  Runs after RoPE.
// ---------------------------------------------------------------------------
__global__ void __launch_bounds__(256)
prep_kv(const float* __restrict__ qkv, _Float16* __restrict__ Kt,
        _Float16* __restrict__ Vg) {
  constexpr int T = 2048, H = 16, R3 = 3072, Dh = 64, HD = 1024;
  __shared__ _Float16 tile[32 * 64];     // [t][d]
  const int tid = threadIdx.x, bidx = blockIdx.x;   // B*H*(T/32) = 2048
  const int tt = bidx & 63;
  const int h  = (bidx >> 6) & 15;
  const int b  = bidx >> 10;
  const int t0 = tt * 32;
  {
    const int r = tid >> 3, c = (tid & 7) * 8;   // r: t-row, c: d-col
    const float* ks = qkv + (size_t)(b * T + t0 + r) * R3 + HD + h * Dh + c;
    const float* vs = ks + HD;
    v8h kk, vv;
#pragma unroll
    for (int i = 0; i < 8; ++i) { kk[i] = (_Float16)ks[i]; vv[i] = (_Float16)vs[i]; }
    *(v8h*)(tile + r * 64 + c) = kk;
    *(v8h*)(Vg + ((size_t)((b * H + h) * T + t0 + r)) * Dh + c) = vv;
  }
  __syncthreads();
  {
    const int d = tid >> 2, c = (tid & 3) * 8;   // d-row, t-col
    v8h ko;
#pragma unroll
    for (int i = 0; i < 8; ++i) ko[i] = tile[(c + i) * 64 + d];
    *(v8h*)(Kt + ((size_t)((b * H + h) * Dh + d)) * T + t0 + c) = ko;
  }
}

// ---------------------------------------------------------------------------
// Barrier-free flash attention.  256 thr = 8 waves; each wave owns 16 q rows.
// K/V B-fragments are per-lane strided global f16 loads (stride T / Dh); next tile is
// prefetched (global_prefetch_b8); LDS only bounces P (C-layout -> A-layout).
// ---------------------------------------------------------------------------
__global__ void __launch_bounds__(256)
attention_wmma(const float* __restrict__ qkv, const _Float16* __restrict__ Kt,
               const _Float16* __restrict__ Vg, float* __restrict__ out) {
  constexpr int T = 2048, H = 16, HD = 1024, R3 = 3072, Dh = 64;
  __shared__ __align__(32) _Float16 Pst[8 * 16 * 32];  // per-wave 16x32 scratch

  const int tid = threadIdx.x, lane = tid & 31, wv = tid >> 5;
  const int bidx = blockIdx.x;          // B*H*(T/128) = 512
  const int qt = bidx & 15;
  const int h  = (bidx >> 4) & 15;
  const int b  = bidx >> 8;
  const int q0 = qt * 128 + wv * 16;
  const float scale = 0.125f;           // D^-0.5, D=64
  const int hi = lane >> 4, nl = lane & 15;

  const float*    qbase = qkv + ((size_t)(b * T + q0)) * R3 + h * Dh;
  const _Float16* kbase = Kt + ((size_t)(b * H + h)) * Dh * T;   // [64][T]
  const _Float16* vbase = Vg + ((size_t)(b * H + h)) * T * Dh;   // [T][64]

  v16h qa0 = load_a_f32(qbase + 0,  R3, lane);
  v16h qa1 = load_a_f32(qbase + 32, R3, lane);

  float mrow[8], lrow[8];
  v8f o[4] = {};
#pragma unroll
  for (int r = 0; r < 8; ++r) { mrow[r] = -1e30f; lrow[r] = 0.0f; }

  _Float16* Pw = Pst + wv * 512;

  for (int j0 = 0; j0 < T; j0 += 32) {
    if (j0 + 32 < T) {  // prefetch next K/V tile rows for this lane
      __builtin_prefetch(kbase + (unsigned)(lane * T) + j0 + 32, 0, 1);
      __builtin_prefetch(kbase + (unsigned)((32 + lane) * T) + j0 + 32, 0, 1);
      __builtin_prefetch(vbase + (unsigned)((j0 + 32 + lane) * Dh), 0, 1);
    }

    // S = Q (16x64) @ K^T (64x32): per-lane strided (stride T) global f16 loads
    v8f s0 = {}, s1 = {};
    {
      v16h kb0 = load_b_f16(kbase + j0, T, lane);
      v16h kb1 = load_b_f16(kbase + j0 + 16, T, lane);
      s0 = WMMA_F16(qa0, kb0, s0);
      s1 = WMMA_F16(qa0, kb1, s1);
      v16h kb2 = load_b_f16(kbase + (unsigned)(32 * T) + j0, T, lane);
      v16h kb3 = load_b_f16(kbase + (unsigned)(32 * T) + j0 + 16, T, lane);
      s0 = WMMA_F16(qa1, kb2, s0);
      s1 = WMMA_F16(qa1, kb3, s1);
    }

    // online softmax (rows live in 16-lane halves of the C fragments)
#pragma unroll
    for (int r = 0; r < 8; ++r) {
      const float v0 = s0[r] * scale;
      const float v1 = s1[r] * scale;
      float rmax = fmaxf(v0, v1);
#pragma unroll
      for (int msk = 1; msk < 16; msk <<= 1)
        rmax = fmaxf(rmax, __shfl_xor(rmax, msk, 32));
      const float mnew = fmaxf(mrow[r], rmax);
      const float corr = __expf(mrow[r] - mnew);
      const float p0 = __expf(v0 - mnew);
      const float p1 = __expf(v1 - mnew);
      float rsum = p0 + p1;
#pragma unroll
      for (int msk = 1; msk < 16; msk <<= 1)
        rsum += __shfl_xor(rsum, msk, 32);
      lrow[r] = lrow[r] * corr + rsum;
      mrow[r] = mnew;
#pragma unroll
      for (int f = 0; f < 4; ++f) o[f][r] *= corr;
      const int m = r + 8 * hi;
      Pw[m * 32 + nl]      = (_Float16)p0;
      Pw[m * 32 + 16 + nl] = (_Float16)p1;
    }

    // O += P (16x32) @ V (32x64)
    v16h pa = load_a_f16(Pw, 32, lane);
#pragma unroll
    for (int f = 0; f < 4; ++f) {
      v16h vb = load_b_f16(vbase + (unsigned)(j0 * Dh) + f * 16, Dh, lane);
      o[f] = WMMA_F16(pa, vb, o[f]);
    }
  }

#pragma unroll
  for (int f = 0; f < 4; ++f)
#pragma unroll
    for (int r = 0; r < 8; ++r) {
      const unsigned row = (unsigned)(q0 + r + 8 * hi);
      const unsigned col = (unsigned)(h * Dh + f * 16 + nl);
      out[(unsigned)(b * T) * HD + row * HD + col] = o[f][r] / lrow[r];
    }
}

// ---------------------------------------------------------------------------
extern "C" void kernel_launch(void* const* d_in, const int* in_sizes, int n_in,
                              void* d_out, int out_size, void* d_ws, size_t ws_size,
                              hipStream_t stream) {
  (void)in_sizes; (void)n_in; (void)out_size; (void)ws_size;
  const float* x      = (const float*)d_in[0];
  const float* w_qkv  = (const float*)d_in[1];
  const float* b_qkv  = (const float*)d_in[2];
  const float* w_proj = (const float*)d_in[3];
  const float* b_proj = (const float*)d_in[4];
  const float* w1     = (const float*)d_in[5];
  const float* b1     = (const float*)d_in[6];
  const float* w2     = (const float*)d_in[7];
  const float* b2     = (const float*)d_in[8];
  const float* w3     = (const float*)d_in[9];
  const float* b3     = (const float*)d_in[10];
  const float* cs     = (const float*)d_in[11];
  const float* sn     = (const float*)d_in[12];
  float* out = (float*)d_out;

  // Workspace (fp32 region, then f16 region).  h reuses dead qkv+attn.
  float* ws   = (float*)d_ws;
  float* qkv  = ws;                                // 4096*3072 f32
  float* attn = qkv + (size_t)4096 * 3072;         // 4096*1024 f32
  float* a    = attn + (size_t)4096 * 1024;        // 4096*1024 f32
  float* hbuf = ws;                                // 4096*4096 f32 (reuse)
  _Float16* f16base = (_Float16*)(a + (size_t)4096 * 1024);
  _Float16* wqkv16  = f16base;                         // [1024][3072]
  _Float16* wproj16 = wqkv16  + (size_t)1024 * 3072;   // [1024][1024]
  _Float16* w1t16   = wproj16 + (size_t)1024 * 1024;   // [1024][4096]
  _Float16* w2t16   = w1t16   + (size_t)1024 * 4096;   // [1024][4096]
  _Float16* w3t16   = w2t16   + (size_t)1024 * 4096;   // [4096][1024]
  _Float16* kt16    = w3t16   + (size_t)4096 * 1024;   // [2][16][64][2048]
  _Float16* vg16    = kt16    + (size_t)2 * 16 * 64 * 2048;

  const int M = 2 * 2048;  // 4096

  // 0) weight convert+transpose to f16 k-major (once per launch)
  transpose_w_f16<<<dim3(1024 / 32, 3072 / 32), 256, 0, stream>>>(w_qkv, wqkv16, 3072, 1024);
  transpose_w_f16<<<dim3(1024 / 32, 1024 / 32), 256, 0, stream>>>(w_proj, wproj16, 1024, 1024);
  transpose_w_f16<<<dim3(1024 / 32, 4096 / 32), 256, 0, stream>>>(w1, w1t16, 4096, 1024);
  transpose_w_f16<<<dim3(1024 / 32, 4096 / 32), 256, 0, stream>>>(w2, w2t16, 4096, 1024);
  transpose_w_f16<<<dim3(4096 / 32, 1024 / 32), 256, 0, stream>>>(w3, w3t16, 1024, 4096);

  // 1) qkv = x @ w_qkv^T + b_qkv
  gemm_bias_wmma<<<dim3(3072 / 64, M / 128), 256, 0, stream>>>(
      x, wqkv16, b_qkv, qkv, M, 3072, 1024);

  // 2) RoPE in place on q,k
  rope_kernel<<<(2 * 2048 * 16 + 255) / 256, 256, 0, stream>>>(qkv, cs, sn);

  // 3) K/V -> f16 (K transposed)
  prep_kv<<<2 * 16 * (2048 / 32), 256, 0, stream>>>(qkv, kt16, vg16);

  // 4) flash attention -> attn [B,T,H*D]
  attention_wmma<<<2 * 16 * (2048 / 128), 256, 0, stream>>>(qkv, kt16, vg16, attn);

  // 5) a = attn @ w_proj^T + b_proj
  gemm_bias_wmma<<<dim3(1024 / 64, M / 128), 256, 0, stream>>>(
      attn, wproj16, b_proj, a, M, 1024, 1024);

  // 6) h = silu(a@w1^T+b1) * (a@w2^T+b2)
  swiglu_wmma<<<dim3(4096 / 64, M / 128), 256, 0, stream>>>(
      a, w1t16, b1, w2t16, b2, hbuf, M, 4096, 1024);

  // 7) out = h @ w3^T + b3
  gemm_bias_wmma<<<dim3(1024 / 64, M / 128), 256, 0, stream>>>(
      hbuf, w3t16, b3, out, M, 1024, 4096);
}
